// FASTMultiHeadAttention_28913719836846
// MI455X (gfx1250) — matrix operand order repaired and checked by reading, not compiled, not run on hardware
//
#include <hip/hip_runtime.h>

#define HH 16
#define NN 2048
#define DD 64

typedef __attribute__((ext_vector_type(16))) _Float16 v16h;
typedef __attribute__((ext_vector_type(8)))  _Float16 v8h;
typedef __attribute__((ext_vector_type(2)))  _Float16 v2h;
typedef __attribute__((ext_vector_type(8)))  float    v8f;

static __device__ __forceinline__ v8f wmma_f16(v16h a, v16h b, v8f c) {
  // D = A(16x32 f16) * B(32x16 f16) + C(16x16 f32)
  return __builtin_amdgcn_wmma_f32_16x16x32_f16(false, a, false, b, (short)0, c,
                                                false, false);
}

// 16 contiguous halves (32B, aligned) -> one B-operand register block.
static __device__ __forceinline__ v16h load16h(const _Float16* p) {
  return *(const v16h*)p;
}
// Two 8-half chunks -> A-operand (callers pass p1 = p0 + 8: contiguous k = 16*hi + s).
static __device__ __forceinline__ v16h load8h8h(const _Float16* p0,
                                                const _Float16* p1) {
  v8h a = *(const v8h*)p0;
  v8h b = *(const v8h*)p1;
  v16h r;
#pragma unroll
  for (int i = 0; i < 8; ++i) { r[i] = a[i]; r[8 + i] = b[i]; }
  return r;
}

// ---------------- pre-pass: f32 -> f16 (+ V transpose) into workspace -------
// ws layout (halves): qw[QK] | kw[QK] | vt[QK] (per head: [d][n]) | rw[(2N-1)*D]
// Each thread converts a pair of elements (packed 32-bit f16 stores).
__global__ __launch_bounds__(256) void convert_f16(
    const float* __restrict__ q, const float* __restrict__ k,
    const float* __restrict__ v, const float* __restrict__ rpe,
    _Float16* __restrict__ ws) {
  const int QK = HH * NN * DD;  // 2097152
  _Float16* qw = ws;
  _Float16* kw = ws + QK;
  _Float16* vt = ws + 2 * QK;
  _Float16* rw = ws + 3 * QK;

  const int i = (blockIdx.x * 256 + threadIdx.x) * 2;
  if (i < QK) {
    v2h t;
    t[0] = (_Float16)q[i]; t[1] = (_Float16)q[i + 1];
    *(v2h*)(qw + i) = t;
    t[0] = (_Float16)k[i]; t[1] = (_Float16)k[i + 1];
    *(v2h*)(kw + i) = t;
    // vt[h][d][n] = v[h][n][d]; pair shares d (n even, NN even)
    const int h = i >> 17;             // / (DD*NN)
    const int rem = i & (DD * NN - 1);
    const int d = rem >> 11;           // / NN
    const int n = rem & (NN - 1);
    const float* vs = v + h * NN * DD + n * DD + d;
    t[0] = (_Float16)vs[0]; t[1] = (_Float16)vs[DD];
    *(v2h*)(vt + i) = t;
  }
  if (i < (2 * NN - 1) * DD) {
    v2h t;
    t[0] = (_Float16)rpe[i]; t[1] = (_Float16)rpe[i + 1];
    *(v2h*)(rw + i) = t;
  }
}

// ---------------- main kernel: 1 wave = (head, 32-row block) -----------------
// min 4 waves/EU -> cap at 256 VGPRs (no s_set_vgpr_msb churn, +occupancy).
__global__ __launch_bounds__(128, 4) void fastmax_fwd(
    const _Float16* __restrict__ ws, float* __restrict__ out) {
  const int QK = HH * NN * DD;
  const _Float16* qw = ws;
  const _Float16* kw = ws + QK;
  const _Float16* vt = ws + 2 * QK;
  const _Float16* rw = ws + 3 * QK;

  // Wave-private LDS (no block barriers needed).
  // P rings stored slot-interleaved per row: element (slot s, row M, col c)
  // at [M*32 + s*16 + c]  ->  row M's 31-wide bias window is contiguous.
  __shared__ float    sPA[4][512];
  __shared__ float    sPB[4][512];
  __shared__ _Float16 sA[4][1024];   // 'a' staging: [rowtile][coltile t][256]

  const int lane  = threadIdx.x & 31;
  const int wave  = threadIdx.x >> 5;
  const int lin   = blockIdx.x * 4 + wave;  // 0..1023
  const int head  = lin >> 6;               // 64 row-pairs per head
  const int bi2   = lin & 63;
  const int biA   = 2 * bi2;
  const int biB   = biA + 1;
  const int laneN = lane & 15;
  const int hi    = lane >> 4;

  const _Float16* qh  = qw + head * (NN * DD);
  const _Float16* kh  = kw + head * (NN * DD);
  const _Float16* vth = vt + head * (DD * NN);
  float*          oh  = out + head * (NN * DD);

  float*    PA   = sPA[wave];
  float*    PB   = sPB[wave];
  _Float16* Astg = sA[wave];

  // Q tiles as A-operands (d 0..31, 32..63) for both row tiles.
  const _Float16* qrA = qh + (biA * 16 + laneN) * DD;
  const _Float16* qrB = qh + (biB * 16 + laneN) * DD;
  const v16h qaA0 = load8h8h(qrA + hi * 16, qrA + 8 + hi * 16);
  const v16h qaA1 = load8h8h(qrA + 32 + hi * 16, qrA + 40 + hi * 16);
  const v16h qaB0 = load8h8h(qrB + hi * 16, qrB + 8 + hi * 16);
  const v16h qaB1 = load8h8h(qrB + 32 + hi * 16, qrB + 40 + hi * 16);

  v8f oA[4] = {};
  v8f oB[4] = {};
  float dA[8], dB[8];
#pragma unroll
  for (int r = 0; r < 8; ++r) { dA[r] = 0.0f; dB[r] = 0.0f; }

  // W(m) = rpe rows [s0+16m, s0+16m+15]; row-tile B's window is W(m-1).
  const int s0 = NN - 16 * (biA + 1);

  auto loadW = [&](int m, v16h& b0, v16h& b1) {
    const _Float16* rr = rw + (s0 + 16 * m + laneN) * DD + hi * 16;
    b0 = load16h(rr);
    b1 = load16h(rr + 32);
  };
  auto storeC = [&](float* ring, int slot, v8f c) {
#pragma unroll
    for (int r = 0; r < 8; ++r)
      ring[(r + 8 * hi) * 32 + slot * 16 + laneN] = c[r];
  };

  // Prologue: PA holds W(0) in slot0; PB holds W(0) slot0, W(-1) slot1.
  {
    v16h b0, b1;
    loadW(0, b0, b1);
    v8f p = {};  p = wmma_f16(qaA0, b0, p);  p = wmma_f16(qaA1, b1, p);
    storeC(PA, 0, p);
    v8f p2 = {}; p2 = wmma_f16(qaB0, b0, p2); p2 = wmma_f16(qaB1, b1, p2);
    storeC(PB, 0, p2);
    loadW(-1, b0, b1);
    v8f p3 = {}; p3 = wmma_f16(qaB0, b0, p3); p3 = wmma_f16(qaB1, b1, p3);
    storeC(PB, 1, p3);  // slot (-1)&1 == 1
  }

  const int npairs = bi2 + 1;  // col tiles 0..biB (always an even count)
  for (int pr = 0; pr < npairs; ++pr) {
    const int bj = pr * 2;
#pragma unroll
    for (int t = 0; t < 2; ++t) {
      const int tile = bj + t;

      // New RPE window W(tile+1); rpe B-operand shared by both row tiles.
      v16h rb0, rb1;
      loadW(tile + 1, rb0, rb1);
      { // P_A(tile+1) now -> PA ring holds {W(tile), W(tile+1)}
        v8f p = {}; p = wmma_f16(qaA0, rb0, p); p = wmma_f16(qaA1, rb1, p);
        storeC(PA, (tile + 1) & 1, p);
      }

      // Scores: K B-operand loaded once, feeds both row tiles.
      const _Float16* kr = kh + (tile * 16 + laneN) * DD + hi * 16;
      v16h kb0 = load16h(kr);
      v16h kb1 = load16h(kr + 32);
      v8f scA = {}; scA = wmma_f16(qaA0, kb0, scA); scA = wmma_f16(qaA1, kb1, scA);
      v8f scB = {}; scB = wmma_f16(qaB0, kb0, scB); scB = wmma_f16(qaB1, kb1, scB);
      __builtin_amdgcn_wave_barrier();

      // Branch-free Toeplitz bias gather (rotated window) + fastmax poly.
      const int shA = (tile & 1) << 4;        // window rotation for PA
      const int shB = ((tile + 1) & 1) << 4;  // == ((tile-1)&1)<<4 for PB
#pragma unroll
      for (int r = 0; r < 8; ++r) {
        const int M  = r + 8 * hi;
        const int qc = 15 + laneN - M;  // in [0,30]
        float biasA = PA[M * 32 + ((qc + shA) & 31)];
        float s = scA[r] + biasA;
        float a = 1.0f + s + 0.5f * s * s;
        if (tile * 16 + laneN > biA * 16 + M) a = 0.0f;
        dA[r] += a;
        Astg[t * 256 + M * 16 + laneN] = (_Float16)a;

        float biasB = PB[M * 32 + ((qc + shB) & 31)];
        float s2 = scB[r] + biasB;
        float a2 = 1.0f + s2 + 0.5f * s2 * s2;
        if (tile * 16 + laneN > biB * 16 + M) a2 = 0.0f;
        dB[r] += a2;
        Astg[512 + t * 256 + M * 16 + laneN] = (_Float16)a2;
      }

      { // P_B(tile+1) AFTER B's gather (its slot held W(tile-1))
        v8f p = {}; p = wmma_f16(qaB0, rb0, p); p = wmma_f16(qaB1, rb1, p);
        storeC(PB, (tile + 1) & 1, p);
      }
    }
    __builtin_amdgcn_wave_barrier();

    // A@V: A-operands from staged 'a'; B-operand = 16 contiguous halves of
    // pre-transposed V (shared by both row tiles).
    v16h aopA = load8h8h(Astg + hi * 256 + laneN * 16,
                         Astg + hi * 256 + laneN * 16 + 8);
    v16h aopB = load8h8h(Astg + 512 + hi * 256 + laneN * 16,
                         Astg + 512 + hi * 256 + laneN * 16 + 8);
#pragma unroll
    for (int n0 = 0; n0 < 4; ++n0) {
      const _Float16* vp = vth + (n0 * 16 + laneN) * NN + bj * 16 + hi * 16;
      v16h bop = load16h(vp);
      oA[n0] = wmma_f16(aopA, bop, oA[n0]);
      oB[n0] = wmma_f16(aopB, bop, oB[n0]);
    }
  }

  // Row denominators: reduce across the 16 lanes sharing each row.
#pragma unroll
  for (int r = 0; r < 8; ++r) {
    float x = dA[r];
    float y = dB[r];
#pragma unroll
    for (int off = 1; off < 16; off <<= 1) {
      x += __shfl_xor(x, off, 32);
      y += __shfl_xor(y, off, 32);
    }
    dA[r] = 1.0f / x;
    dB[r] = 1.0f / y;
  }

  // Write O (C-layout -> coalesced 16-wide stores).
#pragma unroll
  for (int n0 = 0; n0 < 4; ++n0) {
#pragma unroll
    for (int r = 0; r < 8; ++r) {
      const int M = r + 8 * hi;
      oh[(biA * 16 + M) * DD + n0 * 16 + laneN] = oA[n0][r] * dA[r];
      oh[(biB * 16 + M) * DD + n0 * 16 + laneN] = oB[n0][r] * dB[r];
    }
  }
}

extern "C" void kernel_launch(void* const* d_in, const int* in_sizes, int n_in,
                              void* d_out, int out_size, void* d_ws, size_t ws_size,
                              hipStream_t stream) {
  const float* q   = (const float*)d_in[0];
  const float* k   = (const float*)d_in[1];
  const float* v   = (const float*)d_in[2];
  // d_in[3] = drop_noise, unused (DROPOUT == 0.0)
  const float* rpe = (const float*)d_in[4];
  float* out = (float*)d_out;
  _Float16* ws = (_Float16*)d_ws;  // needs 3*QK + (2N-1)*D halves ~= 12.5 MB

  // Pre-pass: f32 -> f16 (+ V transpose) once per launch.
  const int QK = HH * NN * DD;
  hipLaunchKernelGGL(convert_f16, dim3(QK / 512), dim3(256), 0, stream,
                     q, k, v, rpe, ws);
  // Main compute: 1024 waves (16 heads x 64 row-pairs), 4 waves/block.
  hipLaunchKernelGGL(fastmax_fwd, dim3(256), dim3(128), 0, stream, ws, out);
}
